// HeCo_18803366822483
// MI455X (gfx1250) — matrix operand order repaired and checked by reading, not compiled, not run on hardware
//
#include <hip/hip_runtime.h>
#include <hip/hip_bf16.h>
#include <math.h>

// ---------------- problem constants (match reference) ----------------
#define N_P   8000
#define N_A   20000
#define N_S   2000
#define E_AP  256000
#define E_SP  16000
#define E_PG  64000
#define DIN   128
#define HDIM  64
#define TAUF  0.8f
#define LAMF  0.5f
#define EPSF  1e-8f
#define SLOPEF 0.01f
#define NTILE (N_P/16)          // 500 tiles per dim
// sim kernel blocking: block = 4 waves stacked vertically (64 rows),
// each wave sweeps 4 column tiles (64 cols). Grid 125x125 covers 8000x8000 exactly.
#define SIM_BLK_WAVES 4
#define SIM_COL_TILES 4
#define SIM_GRID (N_P / 64)     // 125

typedef __attribute__((ext_vector_type(16))) _Float16 v16h;
typedef __attribute__((ext_vector_type(8)))  _Float16 v8h;
typedef __attribute__((ext_vector_type(8)))  float    v8f;

// ---------------- utility kernels ----------------
__global__ void k_fill(float* p, float v, int n) {
  int i = blockIdx.x * blockDim.x + threadIdx.x;
  if (i < n) p[i] = v;
}

// Y[n,j] = act(b[j] + sum_k X[n,k]*W[j,k]); act: 0=none, 1=elu
__global__ void k_lin(const float* __restrict__ X, const float* __restrict__ W,
                      const float* __restrict__ b, float* __restrict__ Y,
                      int M, int K, int Ho, int act) {
  int i = blockIdx.x * blockDim.x + threadIdx.x;
  if (i >= M * Ho) return;
  int n = i / Ho, j = i - n * Ho;
  const float* x = X + (size_t)n * K;
  const float* w = W + (size_t)j * K;
  float acc = b[j];
  for (int k = 0; k < K; ++k) acc = fmaf(x[k], w[k], acc);
  if (act == 1) acc = acc > 0.f ? acc : (__expf(acc) - 1.f);
  Y[i] = acc;
}

// out[n] = sum_k X[n,k]*v[k]
__global__ void k_dotvec(const float* __restrict__ X, const float* __restrict__ v,
                         float* __restrict__ out, int M) {
  int n = blockIdx.x * blockDim.x + threadIdx.x;
  if (n >= M) return;
  const float* x = X + (size_t)n * HDIM;
  float a = 0.f;
  for (int k = 0; k < HDIM; ++k) a = fmaf(x[k], v[k], a);
  out[n] = a;
}

__device__ inline void atomicMaxF(float* addr, float val) {
  if (val >= 0.f) atomicMax((int*)addr, __float_as_int(val));
  else            atomicMin((unsigned int*)addr, __float_as_uint(val));
}

// ---------------- GAT (edge softmax) ----------------
__global__ void k_edge_max(const int* __restrict__ src, const int* __restrict__ dst,
                           const float* __restrict__ el, const float* __restrict__ er,
                           float* __restrict__ m, int E) {
  int i = blockIdx.x * blockDim.x + threadIdx.x;
  if (i >= E) return;
  float e = el[src[i]] + er[dst[i]];
  e = e > 0.f ? e : SLOPEF * e;
  atomicMaxF(&m[dst[i]], e);
}

__global__ void k_edge_exp(const int* __restrict__ src, const int* __restrict__ dst,
                           const float* __restrict__ el, const float* __restrict__ er,
                           const float* __restrict__ m, float* __restrict__ ex,
                           float* __restrict__ s, int E) {
  int i = blockIdx.x * blockDim.x + threadIdx.x;
  if (i >= E) return;
  int d = dst[i];
  float e = el[src[i]] + er[d];
  e = e > 0.f ? e : SLOPEF * e;
  float x = __expf(e - m[d]);
  ex[i] = x;
  atomicAdd(&s[d], x);
}

// per (edge,h): z[dst,h] += feat[src,h]*ex[e]
__global__ void k_edge_scatter(const int* __restrict__ src, const int* __restrict__ dst,
                               const float* __restrict__ feat, const float* __restrict__ ex,
                               float* __restrict__ z, int E) {
  int i = blockIdx.x * blockDim.x + threadIdx.x;
  if (i >= E * HDIM) return;
  int e = i >> 6, h = i & 63;
  atomicAdd(&z[(size_t)dst[e] * HDIM + h], feat[(size_t)src[e] * HDIM + h] * ex[e]);
}

__global__ void k_gat_final(float* __restrict__ z, const float* __restrict__ s, int M) {
  int i = blockIdx.x * blockDim.x + threadIdx.x;
  if (i >= M * HDIM) return;
  float sv = s[i >> 6];
  float v = sv > 0.f ? z[i] / sv : 0.f;
  z[i] = v > 0.f ? v : (__expf(v) - 1.f);   // ELU
}

// ---------------- meta-path GraphConv ----------------
__global__ void k_pg_scatter(const int* __restrict__ src, const int* __restrict__ dst,
                             const float* __restrict__ feat, float* __restrict__ agg, int E) {
  int i = blockIdx.x * blockDim.x + threadIdx.x;
  if (i >= E * HDIM) return;
  int e = i >> 6, h = i & 63;
  atomicAdd(&agg[(size_t)dst[e] * HDIM + h], feat[(size_t)src[e] * HDIM + h]);
}

__global__ void k_pg_deg(const int* __restrict__ dst, float* __restrict__ deg, int E) {
  int i = blockIdx.x * blockDim.x + threadIdx.x;
  if (i < E) atomicAdd(&deg[dst[i]], 1.f);
}

// rst[n,j] = PReLU( b_g[j] + sum_k (agg[n,k]/max(deg,1)) * W_g[k,j] )
__global__ void k_graphconv(const float* __restrict__ agg, const float* __restrict__ deg,
                            const float* __restrict__ Wg, const float* __restrict__ bg,
                            const float* __restrict__ prelu_a, float* __restrict__ out) {
  int i = blockIdx.x * blockDim.x + threadIdx.x;
  if (i >= N_P * HDIM) return;
  int n = i >> 6, j = i & 63;
  float invd = 1.f / fmaxf(deg[n], 1.f);
  const float* a = agg + (size_t)n * HDIM;
  float acc = bg[j];
  for (int k = 0; k < HDIM; ++k) acc = fmaf(a[k] * invd, Wg[k * HDIM + j], acc);
  float al = *prelu_a;
  out[i] = acc > 0.f ? acc : al * acc;
}

// ---------------- semantic attention ----------------
__global__ void k_sem_score(const float* __restrict__ z_ap, const float* __restrict__ z_sp,
                            const float* __restrict__ Wsem, const float* __restrict__ bsem,
                            const float* __restrict__ asem, float* __restrict__ wscore) {
  int i = blockIdx.x * blockDim.x + threadIdx.x;
  if (i >= 2 * N_P) return;
  int m = (i >= N_P) ? 1 : 0;
  int n = m ? (i - N_P) : i;
  const float* z = (m ? z_sp : z_ap) + (size_t)n * HDIM;
  float sc = 0.f;
  for (int j = 0; j < HDIM; ++j) {
    float t = bsem[j];
    for (int k = 0; k < HDIM; ++k) t = fmaf(Wsem[j * HDIM + k], z[k], t);
    sc = fmaf(tanhf(t), asem[j], sc);
  }
  atomicAdd(&wscore[m], sc * (1.0f / (float)N_P));
}

__global__ void k_sem_combine(const float* __restrict__ z_ap, const float* __restrict__ z_sp,
                              const float* __restrict__ wscore, float* __restrict__ z_sc) {
  int i = blockIdx.x * blockDim.x + threadIdx.x;
  if (i >= N_P * HDIM) return;
  float w0 = wscore[0], w1 = wscore[1];
  float mx = fmaxf(w0, w1);
  float e0 = __expf(w0 - mx), e1 = __expf(w1 - mx);
  float inv = 1.f / (e0 + e1);
  z_sc[i] = (e0 * inv) * z_ap[i] + (e1 * inv) * z_sp[i];
}

// ---------------- row-normalize to f16 ----------------
__global__ void k_norm_half(const float* __restrict__ z, unsigned short* __restrict__ outh, int M) {
  int n = blockIdx.x * blockDim.x + threadIdx.x;
  if (n >= M) return;
  const float* x = z + (size_t)n * HDIM;
  float ss = 0.f;
  for (int k = 0; k < HDIM; ++k) ss = fmaf(x[k], x[k], ss);
  float inv = 1.f / sqrtf(fmaxf(ss, 1e-30f));
  _Float16* o = reinterpret_cast<_Float16*>(outh) + (size_t)n * HDIM;
  for (int k = 0; k < HDIM; ++k) o[k] = (_Float16)(x[k] * inv);
}

// ---------------- fused WMMA similarity + masked row/col reductions ----------------
// sim[i,j] = exp(dot(zsn_i, zmn_j)/tau), never materialized (saves 2x256MB traffic).
// Blocking to cut global atomics 4x (16M -> 4M):
//   wave  = one 16-row tile x 4 column tiles (A fragments reused 4x; row partials
//           accumulate in registers across the 4 col tiles -> row atomics /4)
//   block = 4 waves stacked vertically (64x64 elements); column partials are
//           combined across waves in LDS -> col atomics /4.
// Grid 125x125 exactly tiles 8000x8000: no guards, EXEC all-1s at every WMMA.
__launch_bounds__(128)
__global__ void k_sim_wmma(const unsigned short* __restrict__ zsn_u,
                           const unsigned short* __restrict__ zmn_u,
                           const int* __restrict__ pos,
                           float* __restrict__ rowsum, float* __restrict__ posrow,
                           float* __restrict__ colsum, float* __restrict__ poscol) {
  const _Float16* zsn = reinterpret_cast<const _Float16*>(zsn_u);
  const _Float16* zmn = reinterpret_cast<const _Float16*>(zmn_u);
  __shared__ float shc[SIM_BLK_WAVES][SIM_COL_TILES][16][2];

  const int lane = threadIdx.x & 31;
  const int wave = threadIdx.x >> 5;                 // 0..3
  const int half = lane >> 4;                        // 0: lanes 0-15, 1: lanes 16-31
  const int r16  = lane & 15;

  const int rowBase  = (blockIdx.y * SIM_BLK_WAVES + wave) * 16;  // this wave's 16 rows
  const int colBase0 = blockIdx.x * (SIM_COL_TILES * 16);         // block's 64 columns

  // A fragments for this wave's row tile (layout per ISA 16-bit A 16x32 table),
  // loaded once, reused across all 4 column tiles.
  const _Float16* arow = zsn + (size_t)(rowBase + r16) * HDIM;
  v16h A0, A1;
  {
    v8h lo0 = *(const v8h*)(arow + 0  + half * 8);
    v8h hi0 = *(const v8h*)(arow + 16 + half * 8);
    A0 = __builtin_shufflevector(lo0, hi0, 0,1,2,3,4,5,6,7,8,9,10,11,12,13,14,15);
    v8h lo1 = *(const v8h*)(arow + 32 + half * 8);
    v8h hi1 = *(const v8h*)(arow + 48 + half * 8);
    A1 = __builtin_shufflevector(lo1, hi1, 0,1,2,3,4,5,6,7,8,9,10,11,12,13,14,15);
  }

  const float invTau = 1.0f / TAUF;
  float rS[8], rP[8];                 // row partials accumulated across the 4 col tiles
#pragma unroll
  for (int r = 0; r < 8; ++r) { rS[r] = 0.f; rP[r] = 0.f; }

#pragma unroll
  for (int j = 0; j < SIM_COL_TILES; ++j) {
    const int colBase = colBase0 + j * 16;
    // B: 32x16 per K-step; lane n holds column n (= zmn row colBase+n)
    const _Float16* brow = zmn + (size_t)(colBase + r16) * HDIM;
    const int* posp = pos + (size_t)(rowBase + half * 8) * N_P + colBase + r16;
    __builtin_prefetch(posp, 0, 1);   // global_prefetch_b8

    v8h blo0 = *(const v8h*)(brow + 0  + half * 8);
    v8h bhi0 = *(const v8h*)(brow + 16 + half * 8);
    v16h B0 = __builtin_shufflevector(blo0, bhi0, 0,1,2,3,4,5,6,7,8,9,10,11,12,13,14,15);
    v8h blo1 = *(const v8h*)(brow + 32 + half * 8);
    v8h bhi1 = *(const v8h*)(brow + 48 + half * 8);
    v16h B1 = __builtin_shufflevector(blo1, bhi1, 0,1,2,3,4,5,6,7,8,9,10,11,12,13,14,15);
    v8f acc = {};
    acc = __builtin_amdgcn_wmma_f32_16x16x32_f16(false, A0, false, B0, (short)0, acc, false, false);
    acc = __builtin_amdgcn_wmma_f32_16x16x32_f16(false, A1, false, B1, (short)0, acc, false, false);

    float colS = 0.f, colP = 0.f;
#pragma unroll
    for (int r = 0; r < 8; ++r) {
      // C layout: VGPR r -> M = r + 8*half, N = r16
      float s = __expf(acc[r] * invTau);
      float pf = (float)posp[(size_t)r * N_P];
      rS[r] += s; rP[r] += s * pf;
      colS += s; colP += s * pf;
    }
    // combine the two half-wave partials for column colBase+r16, park in LDS
    colS += __shfl_xor(colS, 16, 32);
    colP += __shfl_xor(colP, 16, 32);
    if (half == 0) {
      shc[wave][j][r16][0] = colS;
      shc[wave][j][r16][1] = colP;
    }
  }

  // row sums: reduce across the 16 lanes of each half-wave, one atomic per row
#pragma unroll
  for (int r = 0; r < 8; ++r) {
    float a = rS[r], p = rP[r];
#pragma unroll
    for (int mm = 1; mm < 16; mm <<= 1) {
      a += __shfl_xor(a, mm, 32);
      p += __shfl_xor(p, mm, 32);
    }
    if (r16 == 0) {
      int row = rowBase + half * 8 + r;
      atomicAdd(&rowsum[row], a);
      atomicAdd(&posrow[row], p);
    }
  }

  // column sums: combine the 4 waves' partials in LDS, one atomic per (col, array)
  __syncthreads();
  int t = threadIdx.x;                 // 128 threads = 4 j * 16 c * 2 arrays
  int j = t >> 5, c = (t >> 1) & 15, arr = t & 1;
  float sum = shc[0][j][c][arr] + shc[1][j][c][arr] + shc[2][j][c][arr] + shc[3][j][c][arr];
  float* dst = arr ? poscol : colsum;
  atomicAdd(&dst[colBase0 + j * 16 + c], sum);
}

// ---------------- final loss ----------------
__global__ void k_loss(const float* __restrict__ rowsum, const float* __restrict__ posrow,
                       const float* __restrict__ colsum, const float* __restrict__ poscol,
                       float* __restrict__ out) {
  __shared__ float sh1[256], sh2[256];
  int t = threadIdx.x;
  float p1 = 0.f, p2 = 0.f;
  for (int i = t; i < N_P; i += 256) {
    p1 += logf(posrow[i] / (rowsum[i] + EPSF));
    p2 += logf(poscol[i] / (colsum[i] + EPSF));
  }
  sh1[t] = p1; sh2[t] = p2;
  __syncthreads();
  for (int s = 128; s > 0; s >>= 1) {
    if (t < s) { sh1[t] += sh1[t + s]; sh2[t] += sh2[t + s]; }
    __syncthreads();
  }
  if (t == 0) out[0] = -(LAMF * sh1[0] + (1.f - LAMF) * sh2[0]) / (float)N_P;
}

// ---------------- launcher ----------------
extern "C" void kernel_launch(void* const* d_in, const int* in_sizes, int n_in,
                              void* d_out, int out_size, void* d_ws, size_t ws_size,
                              hipStream_t stream) {
  // inputs (setup_inputs dict order)
  const float* feat_paper   = (const float*)d_in[0];
  const float* feat_author  = (const float*)d_in[1];
  const float* feat_subject = (const float*)d_in[2];
  const float* pos_feat     = (const float*)d_in[3];
  const int*   ap_src = (const int*)d_in[4];
  const int*   ap_dst = (const int*)d_in[5];
  const int*   sp_src = (const int*)d_in[6];
  const int*   sp_dst = (const int*)d_in[7];
  const int*   pg_src = (const int*)d_in[8];
  const int*   pg_dst = (const int*)d_in[9];
  const int*   pos    = (const int*)d_in[10];
  const float* W_paper = (const float*)d_in[11];  const float* b_paper = (const float*)d_in[12];
  const float* W_author= (const float*)d_in[13];  const float* b_author= (const float*)d_in[14];
  const float* W_subj  = (const float*)d_in[15];  const float* b_subj  = (const float*)d_in[16];
  const float* al_ap = (const float*)d_in[17];    const float* ar_ap = (const float*)d_in[18];
  const float* al_sp = (const float*)d_in[19];    const float* ar_sp = (const float*)d_in[20];
  const float* W_sem = (const float*)d_in[21];    const float* b_sem = (const float*)d_in[22];
  const float* a_sem = (const float*)d_in[23];
  const float* W_g   = (const float*)d_in[24];    const float* b_g   = (const float*)d_in[25];
  const float* prelu_a = (const float*)d_in[26];
  const float* W_p1 = (const float*)d_in[27];     const float* b_p1 = (const float*)d_in[28];
  const float* W_p2 = (const float*)d_in[29];     const float* b_p2 = (const float*)d_in[30];
  const float* W_pred = (const float*)d_in[31];   const float* b_pred = (const float*)d_in[32];

  float* out = (float*)d_out;   // [0]=loss, [1..]=pred (8000x64)

  // workspace partition (64-float aligned regions)
  float* base = (float*)d_ws;
  size_t o = 0;
  auto alloc = [&](size_t n) { float* p = base + o; o += (n + 63) & ~(size_t)63; return p; };
  float* h_p   = alloc((size_t)N_P * HDIM);
  float* h_a   = alloc((size_t)N_A * HDIM);
  float* h_s   = alloc((size_t)N_S * HDIM);
  float* pos_h = alloc((size_t)N_P * HDIM);
  float* z_ap  = alloc((size_t)N_P * HDIM);
  float* z_sp  = alloc((size_t)N_P * HDIM);
  float* z_sc  = alloc((size_t)N_P * HDIM);
  float* agg   = alloc((size_t)N_P * HDIM);
  float* z_mp  = alloc((size_t)N_P * HDIM);
  float* tmp   = alloc((size_t)N_P * HDIM);
  float* zs    = alloc((size_t)N_P * HDIM);
  float* zm    = alloc((size_t)N_P * HDIM);
  float* el_ap = alloc(N_A);
  float* er_ap = alloc(N_P);
  float* el_sp = alloc(N_S);
  float* er_sp = alloc(N_P);
  float* m_ap  = alloc(N_P);
  float* s_ap  = alloc(N_P);
  float* m_sp  = alloc(N_P);
  float* s_sp  = alloc(N_P);
  float* ex_ap = alloc(E_AP);
  float* ex_sp = alloc(E_SP);
  float* deg   = alloc(N_P);
  float* rowsum = alloc(N_P);
  float* colsum = alloc(N_P);
  float* posrow = alloc(N_P);
  float* poscol = alloc(N_P);
  float* wscore = alloc(64);
  unsigned short* zsn = (unsigned short*)alloc((size_t)N_P * HDIM / 2);
  unsigned short* zmn = (unsigned short*)alloc((size_t)N_P * HDIM / 2);

  const int B = 256;
  auto G = [](long n) { return (unsigned)((n + 255) / 256); };

  // --- zero / -inf init (required every call: accumulators) ---
  hipLaunchKernelGGL(k_fill, dim3(G((long)N_P*HDIM)), dim3(B), 0, stream, z_ap, 0.f, N_P*HDIM);
  hipLaunchKernelGGL(k_fill, dim3(G((long)N_P*HDIM)), dim3(B), 0, stream, z_sp, 0.f, N_P*HDIM);
  hipLaunchKernelGGL(k_fill, dim3(G((long)N_P*HDIM)), dim3(B), 0, stream, agg,  0.f, N_P*HDIM);
  hipLaunchKernelGGL(k_fill, dim3(G(N_P)), dim3(B), 0, stream, s_ap, 0.f, N_P);
  hipLaunchKernelGGL(k_fill, dim3(G(N_P)), dim3(B), 0, stream, s_sp, 0.f, N_P);
  hipLaunchKernelGGL(k_fill, dim3(G(N_P)), dim3(B), 0, stream, deg,  0.f, N_P);
  hipLaunchKernelGGL(k_fill, dim3(G(N_P)), dim3(B), 0, stream, rowsum, 0.f, N_P);
  hipLaunchKernelGGL(k_fill, dim3(G(N_P)), dim3(B), 0, stream, colsum, 0.f, N_P);
  hipLaunchKernelGGL(k_fill, dim3(G(N_P)), dim3(B), 0, stream, posrow, 0.f, N_P);
  hipLaunchKernelGGL(k_fill, dim3(G(N_P)), dim3(B), 0, stream, poscol, 0.f, N_P);
  hipLaunchKernelGGL(k_fill, dim3(1), dim3(B), 0, stream, wscore, 0.f, 2);
  hipLaunchKernelGGL(k_fill, dim3(G(N_P)), dim3(B), 0, stream, m_ap, -1e30f, N_P);
  hipLaunchKernelGGL(k_fill, dim3(G(N_P)), dim3(B), 0, stream, m_sp, -1e30f, N_P);

  // --- input projections (Linear + ELU) ---
  hipLaunchKernelGGL(k_lin, dim3(G((long)N_P*HDIM)), dim3(B), 0, stream, feat_paper,   W_paper, b_paper, h_p,   N_P, DIN, HDIM, 1);
  hipLaunchKernelGGL(k_lin, dim3(G((long)N_A*HDIM)), dim3(B), 0, stream, feat_author,  W_author,b_author,h_a,   N_A, DIN, HDIM, 1);
  hipLaunchKernelGGL(k_lin, dim3(G((long)N_S*HDIM)), dim3(B), 0, stream, feat_subject, W_subj,  b_subj,  h_s,   N_S, DIN, HDIM, 1);
  hipLaunchKernelGGL(k_lin, dim3(G((long)N_P*HDIM)), dim3(B), 0, stream, pos_feat,     W_paper, b_paper, pos_h, N_P, DIN, HDIM, 1);

  // --- GAT attention scalars ---
  hipLaunchKernelGGL(k_dotvec, dim3(G(N_A)), dim3(B), 0, stream, h_a, al_ap, el_ap, N_A);
  hipLaunchKernelGGL(k_dotvec, dim3(G(N_P)), dim3(B), 0, stream, h_p, ar_ap, er_ap, N_P);
  hipLaunchKernelGGL(k_dotvec, dim3(G(N_S)), dim3(B), 0, stream, h_s, al_sp, el_sp, N_S);
  hipLaunchKernelGGL(k_dotvec, dim3(G(N_P)), dim3(B), 0, stream, h_p, ar_sp, er_sp, N_P);

  // --- GAT A->P ---
  hipLaunchKernelGGL(k_edge_max,     dim3(G(E_AP)), dim3(B), 0, stream, ap_src, ap_dst, el_ap, er_ap, m_ap, E_AP);
  hipLaunchKernelGGL(k_edge_exp,     dim3(G(E_AP)), dim3(B), 0, stream, ap_src, ap_dst, el_ap, er_ap, m_ap, ex_ap, s_ap, E_AP);
  hipLaunchKernelGGL(k_edge_scatter, dim3(G((long)E_AP*HDIM)), dim3(B), 0, stream, ap_src, ap_dst, h_a, ex_ap, z_ap, E_AP);
  hipLaunchKernelGGL(k_gat_final,    dim3(G((long)N_P*HDIM)), dim3(B), 0, stream, z_ap, s_ap, N_P);

  // --- GAT S->P ---
  hipLaunchKernelGGL(k_edge_max,     dim3(G(E_SP)), dim3(B), 0, stream, sp_src, sp_dst, el_sp, er_sp, m_sp, E_SP);
  hipLaunchKernelGGL(k_edge_exp,     dim3(G(E_SP)), dim3(B), 0, stream, sp_src, sp_dst, el_sp, er_sp, m_sp, ex_sp, s_sp, E_SP);
  hipLaunchKernelGGL(k_edge_scatter, dim3(G((long)E_SP*HDIM)), dim3(B), 0, stream, sp_src, sp_dst, h_s, ex_sp, z_sp, E_SP);
  hipLaunchKernelGGL(k_gat_final,    dim3(G((long)N_P*HDIM)), dim3(B), 0, stream, z_sp, s_sp, N_P);

  // --- semantic attention ---
  hipLaunchKernelGGL(k_sem_score,   dim3(G(2L*N_P)), dim3(B), 0, stream, z_ap, z_sp, W_sem, b_sem, a_sem, wscore);
  hipLaunchKernelGGL(k_sem_combine, dim3(G((long)N_P*HDIM)), dim3(B), 0, stream, z_ap, z_sp, wscore, z_sc);

  // --- meta-path GraphConv + PReLU ---
  hipLaunchKernelGGL(k_pg_scatter, dim3(G((long)E_PG*HDIM)), dim3(B), 0, stream, pg_src, pg_dst, pos_h, agg, E_PG);
  hipLaunchKernelGGL(k_pg_deg,     dim3(G(E_PG)), dim3(B), 0, stream, pg_dst, deg, E_PG);
  hipLaunchKernelGGL(k_graphconv,  dim3(G((long)N_P*HDIM)), dim3(B), 0, stream, agg, deg, W_g, b_g, prelu_a, z_mp);

  // --- projection heads ---
  hipLaunchKernelGGL(k_lin, dim3(G((long)N_P*HDIM)), dim3(B), 0, stream, z_sc, W_p1, b_p1, tmp, N_P, HDIM, HDIM, 1);
  hipLaunchKernelGGL(k_lin, dim3(G((long)N_P*HDIM)), dim3(B), 0, stream, tmp,  W_p2, b_p2, zs,  N_P, HDIM, HDIM, 0);
  hipLaunchKernelGGL(k_lin, dim3(G((long)N_P*HDIM)), dim3(B), 0, stream, z_mp, W_p1, b_p1, tmp, N_P, HDIM, HDIM, 1);
  hipLaunchKernelGGL(k_lin, dim3(G((long)N_P*HDIM)), dim3(B), 0, stream, tmp,  W_p2, b_p2, zm,  N_P, HDIM, HDIM, 0);

  // --- normalize rows -> f16 WMMA operand tables (1 MB each, L2-resident, ~500x reuse) ---
  hipLaunchKernelGGL(k_norm_half, dim3(G(N_P)), dim3(B), 0, stream, zs, zsn, N_P);
  hipLaunchKernelGGL(k_norm_half, dim3(G(N_P)), dim3(B), 0, stream, zm, zmn, N_P);

  // --- fused WMMA similarity: 125x125 blocks of 4 waves, each block = 64x64 elements ---
  hipLaunchKernelGGL(k_sim_wmma, dim3(SIM_GRID, SIM_GRID), dim3(128), 0, stream,
                     zsn, zmn, pos, rowsum, posrow, colsum, poscol);

  // --- loss + prediction head ---
  hipLaunchKernelGGL(k_loss, dim3(1), dim3(B), 0, stream, rowsum, posrow, colsum, poscol, out);
  hipLaunchKernelGGL(k_lin,  dim3(G((long)N_P*HDIM)), dim3(B), 0, stream, z_sc, W_pred, b_pred, out + 1, N_P, HDIM, HDIM, 0);
}
